// Encoder_4269197492519
// MI455X (gfx1250) — matrix operand order repaired and checked by reading, not compiled, not run on hardware
//
#include <hip/hip_runtime.h>
#include <hip/hip_bf16.h>

typedef __attribute__((ext_vector_type(16))) _Float16 v16h;
typedef __attribute__((ext_vector_type(8)))  _Float16 v8h;
typedef __attribute__((ext_vector_type(4)))  _Float16 v4h;
typedef __attribute__((ext_vector_type(8)))  float    v8f;

#define D_DIM 128

// ---------------------------------------------------------------------------
// Zero-fill f32 region (sum accumulators + degree counters)
// ---------------------------------------------------------------------------
__global__ void zero_f32(float* __restrict__ p, long long n) {
    long long t = (long long)blockIdx.x * blockDim.x + threadIdx.x;
    if (t < n) p[t] = 0.0f;
}

// ---------------------------------------------------------------------------
// Wt[n][k] = (f16) W[k][n]   -- 128x128 transpose+convert (per weight matrix)
// ---------------------------------------------------------------------------
__global__ void transpose_w_f16(const float* __restrict__ W, _Float16* __restrict__ Wt) {
    int t = blockIdx.x * blockDim.x + threadIdx.x;   // 128*128 threads
    int k = t & 127, n = t >> 7;
    Wt[(size_t)n * D_DIM + k] = (_Float16)W[(size_t)k * D_DIM + n];
}

// ---------------------------------------------------------------------------
// Xh = (f16) X   (vectorized: float4 in, 4 halves out)
// ---------------------------------------------------------------------------
__global__ void cvt_f16(const float* __restrict__ X, _Float16* __restrict__ Xh, long long n4) {
    long long t = (long long)blockIdx.x * blockDim.x + threadIdx.x;
    if (t >= n4) return;
    float4 v = ((const float4*)X)[t];
    v4h h = { (_Float16)v.x, (_Float16)v.y, (_Float16)v.z, (_Float16)v.w };
    ((v4h*)Xh)[t] = h;
}

// ---------------------------------------------------------------------------
// x_user_h[row] = (f16) user_emb[user_ids[row]]   (one wave per row)
// ---------------------------------------------------------------------------
__global__ void gather_rows_h(const float* __restrict__ emb, const int* __restrict__ ids,
                              _Float16* __restrict__ outh, int nrows) {
    int t = blockIdx.x * blockDim.x + threadIdx.x;
    int row = t >> 5, lane = t & 31;
    if (row >= nrows) return;
    int id = ids[row];
    float4 v = ((const float4*)(emb + (size_t)id * D_DIM))[lane];
    v4h h = { (_Float16)v.x, (_Float16)v.y, (_Float16)v.z, (_Float16)v.w };
    ((v4h*)(outh + (size_t)row * D_DIM))[lane] = h;
}

// ---------------------------------------------------------------------------
// Scatter-add: agg[dst] += feat[ ids ? ids[edge0[e]] : edge0[e] ]; cnt[dst]+=1
// One wave per edge; per-lane float4 gather + 4 f32 atomics (L2-resident).
// ---------------------------------------------------------------------------
__global__ void scatter_add_idx(const float* __restrict__ feat, const int* __restrict__ ids,
                                const int* __restrict__ edge, int E,
                                float* __restrict__ agg, float* __restrict__ cnt) {
    long long t = (long long)blockIdx.x * blockDim.x + threadIdx.x;
    int e = (int)(t >> 5), lane = (int)(t & 31);
    if (e >= E) return;
    int s = edge[e];         // row 0: src indices
    int d = edge[E + e];     // row 1: dst indices
    int srow = ids ? ids[s] : s;
    float4 v = ((const float4*)(feat + (size_t)srow * D_DIM))[lane];
    float* dst = agg + (size_t)d * D_DIM + lane * 4;
    atomicAdd(dst + 0, v.x);
    atomicAdd(dst + 1, v.y);
    atomicAdd(dst + 2, v.z);
    atomicAdd(dst + 3, v.w);
    if (lane == 0) atomicAdd(cnt + d, 1.0f);
}

// ---------------------------------------------------------------------------
// aggh[row] = (f16)( agg[row] / max(cnt[row],1) )   -- mean + convert fused
// ---------------------------------------------------------------------------
__global__ void div_rows_h(const float* __restrict__ agg, const float* __restrict__ cnt,
                           _Float16* __restrict__ aggh, int nrows) {
    int t = blockIdx.x * blockDim.x + threadIdx.x;
    int row = t >> 5, lane = t & 31;
    if (row >= nrows) return;
    float inv = 1.0f / fmaxf(cnt[row], 1.0f);
    float4 v = ((const float4*)(agg + (size_t)row * D_DIM))[lane];
    v4h h = { (_Float16)(v.x * inv), (_Float16)(v.y * inv),
              (_Float16)(v.z * inv), (_Float16)(v.w * inv) };
    ((v4h*)(aggh + (size_t)row * D_DIM))[lane] = h;
}

// ---------------------------------------------------------------------------
// WMMA fragment readers (ISA 7.12.2 wave32 layouts)
// A 16x32 f16: lane m=L%16; halves [0..7]=K0+8*hl+0..7, [8..15]=K0+16+8*hl+0..7
//   -> two contiguous 16B runs (from LDS tile, row-major f16)
// ---------------------------------------------------------------------------
__device__ __forceinline__ v16h lds_a_frag(const _Float16* tile, int k0, int lane) {
    const _Float16* base = tile + (lane & 15) * D_DIM + k0 + ((lane & 16) ? 8 : 0);
    v8h lo = *(const v8h*)base;
    v8h hi = *(const v8h*)(base + 16);
    v16h a;
#pragma unroll
    for (int i = 0; i < 8; ++i) { a[i] = lo[i]; a[i + 8] = hi[i]; }
    return a;
}

// B 32x16 f16: lane n=L%16; [0..7]=Wt[n][K0+8*hl+0..7], [8..15]=Wt[n][K0+16+8*hl+0..7]
//   -> two contiguous 16B runs in the transposed f16 weights.
__device__ __forceinline__ v16h ld_b_frag(const _Float16* __restrict__ Wt, int n, int k0, int lane) {
    const _Float16* base = Wt + (size_t)n * D_DIM + k0 + ((lane & 16) ? 8 : 0);
    v8h lo = *(const v8h*)base;
    v8h hi = *(const v8h*)(base + 16);
    v16h b;
#pragma unroll
    for (int i = 0; i < 8; ++i) { b[i] = lo[i]; b[i + 8] = hi[i]; }
    return b;
}

// ---------------------------------------------------------------------------
// Fused dual GEMM: Out = A1 @ W1 + A2 @ W2 + bias   (A: [M,128] f16, Wt: f16^T)
// 256 threads = 8 waves; block owns m-tile, wave w owns n-tile w (N=128).
// A-tiles (2 x 16x128 f16 = 8KB) staged in LDS once; branchless -> EXEC all 1s.
// ---------------------------------------------------------------------------
__global__ __launch_bounds__(256) void gemm_dual_bias(
    const _Float16* __restrict__ A1h, const _Float16* __restrict__ Wt1,
    const _Float16* __restrict__ A2h, const _Float16* __restrict__ Wt2,
    const float* __restrict__ bias, float* __restrict__ Out) {
    __shared__ _Float16 atile[2][16 * D_DIM];

    int t = threadIdx.x;
    int m0 = blockIdx.x * 16;
    {   // cooperative stage: 2 tiles * 2048 halves; each thread copies 32B
        int mtx = t >> 7;
        int off = (t & 127) * 16;
        const uint4* s4 = (const uint4*)((mtx ? A2h : A1h) + (size_t)m0 * D_DIM + off);
        uint4* d4 = (uint4*)&atile[mtx][off];
        d4[0] = s4[0];
        d4[1] = s4[1];
    }
    __syncthreads();

    int lane = t & 31;
    int n = (t >> 5) * 16 + (lane & 15);
    v8f acc = {};
#pragma unroll
    for (int k0 = 0; k0 < D_DIM; k0 += 32) {
        v16h a1 = lds_a_frag(atile[0], k0, lane);
        v16h b1 = ld_b_frag(Wt1, n, k0, lane);
        acc = __builtin_amdgcn_wmma_f32_16x16x32_f16(false, a1, false, b1,
                                                     (short)0, acc, false, false);
        v16h a2 = lds_a_frag(atile[1], k0, lane);
        v16h b2 = ld_b_frag(Wt2, n, k0, lane);
        acc = __builtin_amdgcn_wmma_f32_16x16x32_f16(false, a2, false, b2,
                                                     (short)0, acc, false, false);
    }

    float bv = bias[n];
    int rbase = m0 + ((lane & 16) ? 8 : 0);   // f32 C/D: VGPR r -> row r (+8 hi half)
#pragma unroll
    for (int r = 0; r < 8; ++r) {
        Out[(size_t)(rbase + r) * D_DIM + n] = acc[r] + bv;
    }
}

// ---------------------------------------------------------------------------
// Launch
// ---------------------------------------------------------------------------
extern "C" void kernel_launch(void* const* d_in, const int* in_sizes, int n_in,
                              void* d_out, int out_size, void* d_ws, size_t ws_size,
                              hipStream_t stream) {
    const int*   user_ids = (const int*)  d_in[0];
    const float* item_x   = (const float*)d_in[1];
    const int*   edge_ui  = (const int*)  d_in[2];
    const int*   edge_iu  = (const int*)  d_in[3];
    const float* user_emb = (const float*)d_in[4];
    const float* W_l_ui   = (const float*)d_in[5];
    const float* W_r_ui   = (const float*)d_in[6];
    const float* b_ui     = (const float*)d_in[7];
    const float* W_l_iu   = (const float*)d_in[8];
    const float* W_r_iu   = (const float*)d_in[9];
    const float* b_iu     = (const float*)d_in[10];

    const int N_USER = in_sizes[0];
    const int N_ITEM = in_sizes[1] / D_DIM;
    const int E      = in_sizes[2] / 2;

    // ws layout: [agg_ui f32 | agg_iu f32 | cnt_item | cnt_user |
    //             x_user_h | item_x_h | agg_ui_h | agg_iu_h | Wt x4 (f16)]
    float* agg_ui   = (float*)d_ws;
    float* agg_iu   = agg_ui + (size_t)N_ITEM * D_DIM;
    float* cnt_item = agg_iu + (size_t)N_USER * D_DIM;
    float* cnt_user = cnt_item + N_ITEM;
    _Float16* x_user_h = (_Float16*)(cnt_user + N_USER);
    _Float16* item_x_h = x_user_h + (size_t)N_USER * D_DIM;
    _Float16* agg_ui_h = item_x_h + (size_t)N_ITEM * D_DIM;
    _Float16* agg_iu_h = agg_ui_h + (size_t)N_ITEM * D_DIM;
    _Float16* Wt_l_ui  = agg_iu_h + (size_t)N_USER * D_DIM;
    _Float16* Wt_r_ui  = Wt_l_ui + D_DIM * D_DIM;
    _Float16* Wt_l_iu  = Wt_r_ui + D_DIM * D_DIM;
    _Float16* Wt_r_iu  = Wt_l_iu + D_DIM * D_DIM;

    float* out_user = (float*)d_out;                       // tuple element 0
    float* out_item = out_user + (size_t)N_USER * D_DIM;   // tuple element 1

    // 1) zero sum accumulators + counters (contiguous f32 region)
    long long nz = (long long)(N_ITEM + N_USER) * D_DIM + N_ITEM + N_USER;
    zero_f32<<<(unsigned)((nz + 255) / 256), 256, 0, stream>>>(agg_ui, nz);

    // 2) weight transpose+convert (tiny)
    transpose_w_f16<<<D_DIM * D_DIM / 256, 256, 0, stream>>>(W_l_ui, Wt_l_ui);
    transpose_w_f16<<<D_DIM * D_DIM / 256, 256, 0, stream>>>(W_r_ui, Wt_r_ui);
    transpose_w_f16<<<D_DIM * D_DIM / 256, 256, 0, stream>>>(W_l_iu, Wt_l_iu);
    transpose_w_f16<<<D_DIM * D_DIM / 256, 256, 0, stream>>>(W_r_iu, Wt_r_iu);

    // 3) f16 activation copies for the tensor cores
    {
        long long n4 = (long long)N_ITEM * (D_DIM / 4);
        cvt_f16<<<(unsigned)((n4 + 255) / 256), 256, 0, stream>>>(item_x, item_x_h, n4);
        long long nt = (long long)N_USER * 32;
        gather_rows_h<<<(unsigned)((nt + 255) / 256), 256, 0, stream>>>(user_emb, user_ids, x_user_h, N_USER);
    }

    // 4) scatter-add both edge types (u->i uses double indirection via user_ids)
    {
        long long nt = (long long)E * 32;
        unsigned nb = (unsigned)((nt + 255) / 256);
        scatter_add_idx<<<nb, 256, 0, stream>>>(user_emb, user_ids, edge_ui, E, agg_ui, cnt_item);
        scatter_add_idx<<<nb, 256, 0, stream>>>(item_x, (const int*)nullptr, edge_iu, E, agg_iu, cnt_user);
    }

    // 5) mean + f16 convert (fused)
    {
        long long nti = (long long)N_ITEM * 32;
        long long ntu = (long long)N_USER * 32;
        div_rows_h<<<(unsigned)((nti + 255) / 256), 256, 0, stream>>>(agg_ui, cnt_item, agg_ui_h, N_ITEM);
        div_rows_h<<<(unsigned)((ntu + 255) / 256), 256, 0, stream>>>(agg_iu, cnt_user, agg_iu_h, N_USER);
    }

    // 6) fused dual GEMMs + bias (WMMA f16 in / f32 accumulate)
    gemm_dual_bias<<<N_ITEM / 16, 256, 0, stream>>>(agg_ui_h, Wt_l_ui, item_x_h, Wt_r_ui, b_ui, out_item);
    gemm_dual_bias<<<N_USER / 16, 256, 0, stream>>>(agg_iu_h, Wt_l_iu, x_user_h, Wt_r_iu, b_iu, out_user);
}
